// EnhancedSeq2Seq_56719338111432
// MI455X (gfx1250) — matrix operand order repaired and checked by reading, not compiled, not run on hardware
//
#include <hip/hip_runtime.h>
#include <hip/hip_bf16.h>
#include <math.h>

typedef __attribute__((ext_vector_type(16))) _Float16 v16h;
typedef __attribute__((ext_vector_type(8)))  _Float16 v8h;
typedef __attribute__((ext_vector_type(8)))  float    v8f;

#define Bsz   64
#define SSRC  30
#define STRG  20
#define Edim  64
#define Hdim  128
#define Vdim  32000
#define NTOK  (Bsz*STRG)      // 1280 MoE rows
#define NBLK  32              // persistent grid blocks
#define NTHR  256
#define SLOTS (82*16)         // padded row slots (max 81 tiles + margin)

// ---------------- helpers ----------------

__device__ __forceinline__ float sigf(float x) { return 1.0f / (1.0f + expf(-x)); }

// Software grid barrier: monotonically increasing counter, target = epoch*NBLK.
// Counter is zeroed with hipMemsetAsync before each persistent kernel launch.
__device__ __forceinline__ void grid_barrier(unsigned* cnt, unsigned target) {
  __syncthreads();
  if (threadIdx.x == 0) {
    __threadfence();
    __hip_atomic_fetch_add(cnt, 1u, __ATOMIC_ACQ_REL, __HIP_MEMORY_SCOPE_AGENT);
    while (__hip_atomic_load(cnt, __ATOMIC_ACQUIRE, __HIP_MEMORY_SCOPE_AGENT) < target) {
      __builtin_amdgcn_s_sleep(1);
    }
  }
  __syncthreads();
}

// ---------------- embedding gather ----------------
// xs[s][b][e] = emb[src[b][s]][e]   (time-major), same for trg.
__global__ void embed_kernel(const int* __restrict__ src, const int* __restrict__ trg,
                             const float* __restrict__ emb,
                             float* __restrict__ xs, float* __restrict__ xt) {
  int g = blockIdx.x * NTHR + threadIdx.x;
  const int nSrc = SSRC * Bsz * Edim;
  if (g < nSrc) {
    int e = g & 63; int sb = g >> 6; int s = sb / Bsz; int b = sb % Bsz;
    xs[g] = emb[(size_t)src[b * SSRC + s] * Edim + e];
  } else if (g < nSrc + STRG * Bsz * Edim) {
    int g2 = g - nSrc;
    int e = g2 & 63; int tb = g2 >> 6; int t = tb / Bsz; int b = tb % Bsz;
    xt[g2] = emb[(size_t)trg[b * STRG + t] * Edim + e];
  }
}

// ---------------- persistent LSTM layer (one thread per (b,i) cell) ----------------
__global__ void lstm_layer_kernel(const float* __restrict__ x, int T, int inDim,
                                  const float* __restrict__ Wih, const float* __restrict__ Whh,
                                  const float* __restrict__ bias,
                                  float* __restrict__ y,
                                  float* __restrict__ hA, float* __restrict__ hB,
                                  float* __restrict__ hFin, float* __restrict__ cFin,
                                  unsigned* __restrict__ barCnt) {
  const int g = blockIdx.x * NTHR + threadIdx.x;   // 0..8191
  const int b = g >> 7, i = g & 127;
  unsigned epoch = 0;
  hA[g] = 0.0f;
  float c = 0.0f, h = 0.0f;
  grid_barrier(barCnt, ++epoch * NBLK);

  const float* w0 = Wih + (size_t)i * inDim;
  const float* w1 = Wih + (size_t)(Hdim + i) * inDim;
  const float* w2 = Wih + (size_t)(2 * Hdim + i) * inDim;
  const float* w3 = Wih + (size_t)(3 * Hdim + i) * inDim;
  const float* u0 = Whh + (size_t)i * Hdim;
  const float* u1 = Whh + (size_t)(Hdim + i) * Hdim;
  const float* u2 = Whh + (size_t)(2 * Hdim + i) * Hdim;
  const float* u3 = Whh + (size_t)(3 * Hdim + i) * Hdim;
  const float bi = bias[i], bf = bias[Hdim + i], bg = bias[2 * Hdim + i], bo = bias[3 * Hdim + i];

  for (int t = 0; t < T; ++t) {
    const float* hin  = (t & 1) ? hB : hA;
    float*       hout = (t & 1) ? hA : hB;
    const float* xr = x + ((size_t)t * Bsz + b) * inDim;
    float zi = bi, zf = bf, zg = bg, zo = bo;
    for (int k = 0; k < inDim; ++k) {
      float xv = xr[k];
      zi = fmaf(xv, w0[k], zi); zf = fmaf(xv, w1[k], zf);
      zg = fmaf(xv, w2[k], zg); zo = fmaf(xv, w3[k], zo);
    }
    const float* hr = hin + b * Hdim;
    for (int j = 0; j < Hdim; ++j) {
      float hv = hr[j];
      zi = fmaf(hv, u0[j], zi); zf = fmaf(hv, u1[j], zf);
      zg = fmaf(hv, u2[j], zg); zo = fmaf(hv, u3[j], zo);
    }
    c = sigf(zf) * c + sigf(zi) * tanhf(zg);
    h = sigf(zo) * tanhf(c);
    y[((size_t)t * Bsz + b) * Hdim + i] = h;
    hout[g] = h;
    grid_barrier(barCnt, ++epoch * NBLK);
  }
  hFin[g] = h;
  cFin[g] = c;
}

// ---------------- attention projection: enc_proj[s,b,d] = encOut[s,b,:]·att_W[d,:] + att_b[d] ----
__global__ void enc_proj_kernel(const float* __restrict__ encOut,
                                const float* __restrict__ attW, const float* __restrict__ attB,
                                float* __restrict__ encProj) {
  int g = blockIdx.x * NTHR + threadIdx.x;
  if (g >= SSRC * Bsz * Hdim) return;
  int sb = g >> 7, d = g & 127;
  const float* er = encOut + (size_t)sb * Hdim;
  const float* wr = attW + (size_t)d * Hdim;
  float acc = attB[d];
  for (int h2 = 0; h2 < Hdim; ++h2) acc = fmaf(er[h2], wr[h2], acc);
  encProj[g] = acc;
}

// ---------------- persistent decoder: attention + 2 LSTM cells per step ----------------
__global__ void decoder_kernel(const float* __restrict__ et,
                               const float* __restrict__ encOut, const float* __restrict__ encProj,
                               const float* __restrict__ attV,
                               const float* __restrict__ Wih0, const float* __restrict__ Whh0, const float* __restrict__ b0,
                               const float* __restrict__ Wih1, const float* __restrict__ Whh1, const float* __restrict__ b1,
                               const float* __restrict__ h0i, const float* __restrict__ c0i,
                               const float* __restrict__ h1i, const float* __restrict__ c1i,
                               float* __restrict__ h0A, float* __restrict__ h0B,
                               float* __restrict__ h1A, float* __restrict__ h1B,
                               float* __restrict__ ctx, float* __restrict__ scores, float* __restrict__ wbuf,
                               float* __restrict__ ys, unsigned* __restrict__ barCnt) {
  const int g = blockIdx.x * NTHR + threadIdx.x;
  const int b = g >> 7, i = g & 127;
  unsigned epoch = 0;
  h0A[g] = h0i[g];
  h1A[g] = h1i[g];
  float c0 = c0i[g], c1 = c1i[g];
  grid_barrier(barCnt, ++epoch * NBLK);

  const float* p00 = Wih0 + (size_t)i * 192;
  const float* p01 = Wih0 + (size_t)(128 + i) * 192;
  const float* p02 = Wih0 + (size_t)(256 + i) * 192;
  const float* p03 = Wih0 + (size_t)(384 + i) * 192;
  const float* q00 = Whh0 + (size_t)i * 128;
  const float* q01 = Whh0 + (size_t)(128 + i) * 128;
  const float* q02 = Whh0 + (size_t)(256 + i) * 128;
  const float* q03 = Whh0 + (size_t)(384 + i) * 128;
  const float* p10 = Wih1 + (size_t)i * 128;
  const float* p11 = Wih1 + (size_t)(128 + i) * 128;
  const float* p12 = Wih1 + (size_t)(256 + i) * 128;
  const float* p13 = Wih1 + (size_t)(384 + i) * 128;
  const float* q10 = Whh1 + (size_t)i * 128;
  const float* q11 = Whh1 + (size_t)(128 + i) * 128;
  const float* q12 = Whh1 + (size_t)(256 + i) * 128;
  const float* q13 = Whh1 + (size_t)(384 + i) * 128;
  const float b0i_ = b0[i], b0f_ = b0[128 + i], b0g_ = b0[256 + i], b0o_ = b0[384 + i];
  const float b1i_ = b1[i], b1f_ = b1[128 + i], b1g_ = b1[256 + i], b1o_ = b1[384 + i];

  for (int t = 0; t < STRG; ++t) {
    const float* h0in = (t & 1) ? h0B : h0A;  float* h0out = (t & 1) ? h0A : h0B;
    const float* h1in = (t & 1) ? h1B : h1A;  float* h1out = (t & 1) ? h1A : h1B;

    // P0: attention scores[b,s] = sum_d tanh(enc_proj + h1) * att_v
    if (g < Bsz * SSRC) {
      int bb = g / SSRC, s = g % SSRC;
      const float* ep = encProj + ((size_t)s * Bsz + bb) * Hdim;
      const float* hr = h1in + bb * Hdim;
      float acc = 0.f;
      for (int d = 0; d < Hdim; ++d) acc = fmaf(tanhf(ep[d] + hr[d]), attV[d], acc);
      scores[bb * SSRC + s] = acc;
    }
    grid_barrier(barCnt, ++epoch * NBLK);

    // P1: softmax over s
    if (g < Bsz) {
      float mx = -1e30f;
      for (int s = 0; s < SSRC; ++s) mx = fmaxf(mx, scores[g * SSRC + s]);
      float ex[SSRC]; float sum = 0.f;
      for (int s = 0; s < SSRC; ++s) { float v = expf(scores[g * SSRC + s] - mx); ex[s] = v; sum += v; }
      float inv = 1.0f / sum;
      for (int s = 0; s < SSRC; ++s) wbuf[g * SSRC + s] = ex[s] * inv;
    }
    grid_barrier(barCnt, ++epoch * NBLK);

    // P2: context[b,i] = sum_s w * encOut ; stash into ys[:,128+i]
    {
      float acc = 0.f;
      for (int s = 0; s < SSRC; ++s)
        acc = fmaf(wbuf[b * SSRC + s], encOut[((size_t)s * Bsz + b) * Hdim + i], acc);
      ctx[g] = acc;
      ys[((size_t)t * Bsz + b) * 256 + 128 + i] = acc;
    }
    grid_barrier(barCnt, ++epoch * NBLK);

    // P3: cell0, input = [et | ctx] (192)
    {
      const float* xr = et + ((size_t)t * Bsz + b) * Edim;
      const float* cr = ctx + b * Hdim;
      float zi = b0i_, zf = b0f_, zg = b0g_, zo = b0o_;
      for (int k = 0; k < Edim; ++k) {
        float xv = xr[k];
        zi = fmaf(xv, p00[k], zi); zf = fmaf(xv, p01[k], zf);
        zg = fmaf(xv, p02[k], zg); zo = fmaf(xv, p03[k], zo);
      }
      for (int k = 0; k < Hdim; ++k) {
        float xv = cr[k];
        zi = fmaf(xv, p00[Edim + k], zi); zf = fmaf(xv, p01[Edim + k], zf);
        zg = fmaf(xv, p02[Edim + k], zg); zo = fmaf(xv, p03[Edim + k], zo);
      }
      const float* hr = h0in + b * Hdim;
      for (int j = 0; j < Hdim; ++j) {
        float hv = hr[j];
        zi = fmaf(hv, q00[j], zi); zf = fmaf(hv, q01[j], zf);
        zg = fmaf(hv, q02[j], zg); zo = fmaf(hv, q03[j], zo);
      }
      c0 = sigf(zf) * c0 + sigf(zi) * tanhf(zg);
      h0out[g] = sigf(zo) * tanhf(c0);
    }
    grid_barrier(barCnt, ++epoch * NBLK);

    // P4: cell1
    {
      const float* xr = h0out + b * Hdim;
      const float* hr = h1in + b * Hdim;
      float zi = b1i_, zf = b1f_, zg = b1g_, zo = b1o_;
      for (int k = 0; k < Hdim; ++k) {
        float xv = xr[k];
        zi = fmaf(xv, p10[k], zi); zf = fmaf(xv, p11[k], zf);
        zg = fmaf(xv, p12[k], zg); zo = fmaf(xv, p13[k], zo);
      }
      for (int j = 0; j < Hdim; ++j) {
        float hv = hr[j];
        zi = fmaf(hv, q10[j], zi); zf = fmaf(hv, q11[j], zf);
        zg = fmaf(hv, q12[j], zg); zo = fmaf(hv, q13[j], zo);
      }
      c1 = sigf(zf) * c1 + sigf(zi) * tanhf(zg);
      float h1v = sigf(zo) * tanhf(c1);
      h1out[g] = h1v;
      ys[((size_t)t * Bsz + b) * 256 + i] = h1v;
    }
    grid_barrier(barCnt, ++epoch * NBLK);
  }
}

// ---------------- MoE gate: argmax expert + bin rows by expert (padded to 16) ----------------
__global__ void gate_bin_kernel(const float* __restrict__ ys,
                                const float* __restrict__ gateW, const float* __restrict__ gateB,
                                int* __restrict__ rows, int* __restrict__ counts) {
  __shared__ unsigned sh_n0, off0, off1;
  __shared__ int sh_ti[NTOK];
  if (threadIdx.x == 0) sh_n0 = 0;
  __syncthreads();
  for (int n = threadIdx.x; n < NTOK; n += NTHR) {
    int b = n / STRG, t = n % STRG;
    const float* xr = ys + ((size_t)t * Bsz + b) * 256;
    float g0 = gateB[0], g1 = gateB[1];
    for (int k = 0; k < 256; ++k) {
      float xv = xr[k];
      g0 = fmaf(xv, gateW[k], g0);
      g1 = fmaf(xv, gateW[256 + k], g1);
    }
    int e = (g1 > g0) ? 1 : 0;   // softmax is monotone; top-1 weight is exactly 1.0
    sh_ti[n] = e;
    if (!e) atomicAdd(&sh_n0, 1u);
  }
  __syncthreads();
  unsigned n0 = sh_n0;
  unsigned n0p = (n0 + 15u) & ~15u;
  if (threadIdx.x == 0) {
    off0 = 0; off1 = n0p;
    unsigned n1 = NTOK - n0;
    counts[0] = (int)(n0p >> 4);                       // row-tiles of expert 0
    counts[1] = (int)((n0p >> 4) + ((n1 + 15u) >> 4)); // total row-tiles
  }
  for (int s = threadIdx.x; s < SLOTS; s += NTHR) rows[s] = -1;
  __syncthreads();
  for (int n = threadIdx.x; n < NTOK; n += NTHR) {
    unsigned pos = atomicAdd(sh_ti[n] ? &off1 : &off0, 1u);
    rows[pos] = n;
  }
}

// ---------------- gather activations to f16, zero-padding unused slots ----------------
__global__ void gather_xg_kernel(const float* __restrict__ ys, const int* __restrict__ rows,
                                 _Float16* __restrict__ xg16) {
  int g = blockIdx.x * NTHR + threadIdx.x;
  int slot = g >> 8, k = g & 255;
  if (slot >= SLOTS) return;
  int row = rows[slot];
  float v = 0.0f;
  if (row >= 0) {
    int b = row / STRG, t = row % STRG;
    v = ys[((size_t)t * Bsz + b) * 256 + k];
  }
  xg16[g] = (_Float16)v;
}

// ---------------- exp_W fp32 -> fp16 (once per launch; lives in L2 afterwards) ----------------
__global__ void conv_w16_kernel(const float* __restrict__ expW, _Float16* __restrict__ w16) {
  size_t g = ((size_t)blockIdx.x * NTHR + threadIdx.x) * 4;
#pragma unroll
  for (int j = 0; j < 4; ++j) w16[g + j] = (_Float16)expW[g + j];
}

// ---------------- MoE GEMM via v_wmma_f32_16x16x32_f16 ----------------
// One wave per 16x16 output tile. K = 256 -> 8 chained WMMAs. Bias preloaded in C.
// D rows scattered back to original token rows through the index table.
__global__ void moe_wmma_kernel(const _Float16* __restrict__ xg16,
                                const _Float16* __restrict__ expw16,
                                const float* __restrict__ expB,
                                const int* __restrict__ rows, const int* __restrict__ counts,
                                float* __restrict__ out) {
  const int rt0 = counts[0];
  const int tot = counts[1];
  const int rt = blockIdx.y;
  if (rt >= tot) return;
  const int wave = threadIdx.x >> 5;
  const int lane = threadIdx.x & 31;
  const int ct = blockIdx.x * 8 + wave;
  if (ct >= Vdim / 16) return;

  const int e = (rt < rt0) ? 0 : 1;
  const int colBase = ct * 16;
  const int n = lane & 15;       // column for B/C/D, row (M) for A
  const int khalf = lane >> 4;

  // A fragments: 16x32 f16 per step; lane half-group selects K = 16*khalf+e (contiguous)
  v16h a[8];
  const _Float16* arow = xg16 + ((size_t)(rt * 16) + n) * 256;
#pragma unroll
  for (int ks = 0; ks < 8; ++ks) {
    const _Float16* ap = arow + ks * 32 + khalf * 16;
    v8h lo = *(const v8h*)(ap);
    v8h hi = *(const v8h*)(ap + 8);
    a[ks] = __builtin_shufflevector(lo, hi, 0, 1, 2, 3, 4, 5, 6, 7, 8, 9, 10, 11, 12, 13, 14, 15);
  }

  // C init = bias (same for all 16 rows of the column)
  float bias = expB[(size_t)e * Vdim + colBase + n];
  v8f c;
#pragma unroll
  for (int r = 0; r < 8; ++r) c[r] = bias;

  // B fragments: lane holds 16 contiguous K for its column -> one 32B load per step
  const _Float16* brow = expw16 + ((size_t)e * Vdim + colBase + n) * 256 + khalf * 16;
#pragma unroll
  for (int ks = 0; ks < 8; ++ks) {
    v16h bv = *(const v16h*)(brow + ks * 32);
    c = __builtin_amdgcn_wmma_f32_16x16x32_f16(false, a[ks], false, bv, (short)0, c, false, false);
  }

  // Scatter D: VGPR r -> row r (lanes 0-15) / row 8+r (lanes 16-31)
  const int mbase = (lane < 16) ? 0 : 8;
#pragma unroll
  for (int r = 0; r < 8; ++r) {
    int row = rows[rt * 16 + mbase + r];
    if (row >= 0) out[(size_t)row * Vdim + colBase + n] = c[r];
  }
}

// ---------------- host side ----------------
extern "C" void kernel_launch(void* const* d_in, const int* in_sizes, int n_in,
                              void* d_out, int out_size, void* d_ws, size_t ws_size,
                              hipStream_t stream) {
  (void)in_sizes; (void)n_in; (void)out_size; (void)ws_size;
  const int*   src     = (const int*)  d_in[0];
  const int*   trg     = (const int*)  d_in[1];
  const float* emb     = (const float*)d_in[2];
  const float* encWih0 = (const float*)d_in[3];
  const float* encWhh0 = (const float*)d_in[4];
  const float* encB0   = (const float*)d_in[5];
  const float* encWih1 = (const float*)d_in[6];
  const float* encWhh1 = (const float*)d_in[7];
  const float* encB1   = (const float*)d_in[8];
  const float* decWih0 = (const float*)d_in[9];
  const float* decWhh0 = (const float*)d_in[10];
  const float* decB0   = (const float*)d_in[11];
  const float* decWih1 = (const float*)d_in[12];
  const float* decWhh1 = (const float*)d_in[13];
  const float* decB1   = (const float*)d_in[14];
  const float* attW    = (const float*)d_in[15];
  const float* attB    = (const float*)d_in[16];
  const float* attV    = (const float*)d_in[17];
  const float* gateW   = (const float*)d_in[18];
  const float* gateB   = (const float*)d_in[19];
  const float* expW    = (const float*)d_in[20];
  const float* expB    = (const float*)d_in[21];
  float* out = (float*)d_out;

  // workspace carve-out (~39 MB total)
  char* w = (char*)d_ws;
  size_t o = 0;
  auto take = [&](size_t bytes) -> char* {
    char* p = w + o;
    o = (o + bytes + 255) & ~(size_t)255;
    return p;
  };
  float* xs      = (float*)take((size_t)SSRC * Bsz * Edim * 4);
  float* xt      = (float*)take((size_t)STRG * Bsz * Edim * 4);
  float* y0      = (float*)take((size_t)SSRC * Bsz * Hdim * 4);
  float* encOut  = (float*)take((size_t)SSRC * Bsz * Hdim * 4);
  float* encProj = (float*)take((size_t)SSRC * Bsz * Hdim * 4);
  float* hA      = (float*)take(8192 * 4);
  float* hB      = (float*)take(8192 * 4);
  float* hF0     = (float*)take(8192 * 4);
  float* cF0     = (float*)take(8192 * 4);
  float* hF1     = (float*)take(8192 * 4);
  float* cF1     = (float*)take(8192 * 4);
  float* h0A     = (float*)take(8192 * 4);
  float* h0B     = (float*)take(8192 * 4);
  float* h1A     = (float*)take(8192 * 4);
  float* h1B     = (float*)take(8192 * 4);
  float* ctx     = (float*)take(8192 * 4);
  float* scores  = (float*)take(Bsz * SSRC * 4);
  float* wbuf    = (float*)take(Bsz * SSRC * 4);
  float* ys      = (float*)take((size_t)STRG * Bsz * 256 * 4);
  int*   rows    = (int*)  take(SLOTS * 4);
  int*   counts  = (int*)  take(8 * 4);
  unsigned* barCnt = (unsigned*)take(16 * 4);
  _Float16* xg16   = (_Float16*)take((size_t)SLOTS * 256 * 2);
  _Float16* expw16 = (_Float16*)take((size_t)2 * Vdim * 256 * 2);

  // reset grid-barrier counters (graph-capture safe)
  hipMemsetAsync(barCnt, 0, 64, stream);

  // exp_W -> f16 (independent of everything else)
  conv_w16_kernel<<<16000, NTHR, 0, stream>>>(expW, expw16);

  // embeddings
  embed_kernel<<<800, NTHR, 0, stream>>>(src, trg, emb, xs, xt);

  // encoder
  lstm_layer_kernel<<<NBLK, NTHR, 0, stream>>>(xs, SSRC, Edim, encWih0, encWhh0, encB0,
                                               y0, hA, hB, hF0, cF0, barCnt + 0);
  lstm_layer_kernel<<<NBLK, NTHR, 0, stream>>>(y0, SSRC, Hdim, encWih1, encWhh1, encB1,
                                               encOut, hA, hB, hF1, cF1, barCnt + 1);

  // attention projection
  enc_proj_kernel<<<960, NTHR, 0, stream>>>(encOut, attW, attB, encProj);

  // decoder
  decoder_kernel<<<NBLK, NTHR, 0, stream>>>(xt, encOut, encProj, attV,
                                            decWih0, decWhh0, decB0,
                                            decWih1, decWhh1, decB1,
                                            hF0, cF0, hF1, cF1,
                                            h0A, h0B, h1A, h1B,
                                            ctx, scores, wbuf, ys, barCnt + 2);

  // MoE: gate/bin, gather to f16, WMMA GEMM with row scatter
  gate_bin_kernel<<<1, NTHR, 0, stream>>>(ys, gateW, gateB, rows, counts);
  gather_xg_kernel<<<SLOTS, NTHR, 0, stream>>>(ys, rows, xg16);
  moe_wmma_kernel<<<dim3(Vdim / 16 / 8, 82), NTHR, 0, stream>>>(xg16, expw16, expB,
                                                                rows, counts, out);
}
